// GPT_29643864277565
// MI455X (gfx1250) — matrix operand order repaired and checked by reading, not compiled, not run on hardware
//
#include <hip/hip_runtime.h>
#include <hip/hip_bf16.h>
#include <math.h>

// ---------------------------------------------------------------------------
// GPT forward for gfx1250 (MI455X). All GEMMs use v_wmma_f32_16x16x32_f16
// (f16 inputs staged via LDS in fragment-ready swizzled layout, f32 accum).
// ---------------------------------------------------------------------------

typedef __attribute__((ext_vector_type(16))) _Float16 v16h;
typedef __attribute__((ext_vector_type(2)))  _Float16 v2h;
typedef __attribute__((ext_vector_type(8)))  float    v8f;

#define GB 2
#define GT 2048
#define GD 1024
#define GH 16
#define GVOCAB 8192
#define GM (GB * GT)      // 4096 rows

#define TILE_M 64
#define TILE_N 64
#define TILE_K 32

// ---------------------------------------------------------------------------
// Embedding: x[b,t,:] = tok_emb[idx[b,t]] + pos_emb[0,t,:] + act_emb[act[b,t]]
// ---------------------------------------------------------------------------
__global__ __launch_bounds__(256)
void embed_kernel(const int* __restrict__ idx, const int* __restrict__ act,
                  const float* __restrict__ tok, const float* __restrict__ pos,
                  const float* __restrict__ aemb, float* __restrict__ x)
{
    int bt = blockIdx.x;                 // 0 .. B*T-1
    int t  = bt % GT;
    int id = idx[bt];
    int a  = act[bt];
    const float* tr = tok  + (size_t)id * GD;
    const float* pr = pos  + (size_t)t  * GD;
    const float* ar = aemb + (size_t)a  * GD;
    float* xr = x + (size_t)bt * GD;
    for (int i = threadIdx.x; i < GD; i += 256)
        xr[i] = tr[i] + pr[i] + ar[i];
}

// ---------------------------------------------------------------------------
// LayerNorm: one block (256 threads = 8 wave32) per row.
// ---------------------------------------------------------------------------
__device__ __forceinline__ float wave_sum(float v) {
    for (int s = 16; s > 0; s >>= 1) v += __shfl_xor(v, s, 32);
    return v;
}

__global__ __launch_bounds__(256)
void ln_kernel(const float* __restrict__ x, const float* __restrict__ g,
               const float* __restrict__ b, float* __restrict__ out, int D)
{
    __shared__ float sm[8];
    size_t row = blockIdx.x;
    const float* xr = x + row * (size_t)D;
    float* orow = out + row * (size_t)D;
    int lane = threadIdx.x & 31, wv = threadIdx.x >> 5;

    float s = 0.f;
    for (int i = threadIdx.x; i < D; i += 256) s += xr[i];
    s = wave_sum(s);
    if (lane == 0) sm[wv] = s;
    __syncthreads();
    float mean = 0.f;
    #pragma unroll
    for (int i = 0; i < 8; ++i) mean += sm[i];
    mean /= (float)D;
    __syncthreads();

    float vs = 0.f;
    for (int i = threadIdx.x; i < D; i += 256) {
        float d = xr[i] - mean;
        vs += d * d;
    }
    vs = wave_sum(vs);
    if (lane == 0) sm[wv] = vs;
    __syncthreads();
    float var = 0.f;
    #pragma unroll
    for (int i = 0; i < 8; ++i) var += sm[i];
    var /= (float)D;
    float rstd = rsqrtf(var + 1e-5f);

    for (int i = threadIdx.x; i < D; i += 256)
        orow[i] = (xr[i] - mean) * rstd * g[i] + b[i];
}

// ---------------------------------------------------------------------------
// WMMA GEMM: C[M,N] = act(A[M,K] @ W[K,N] + bias) (+ residual)
// Block = 128 threads (4 waves). C tile 64x64; wave w owns rows w*16..w*16+15.
// K must be a multiple of 32; M a multiple of 64; N bounds-checked.
//
// LDS tiles are stored in *fragment-ready* layout so each lane's 16 f16
// operand values are 32 contiguous bytes (2x ds_load_b128 per fragment):
//   A row r (32 halves): natural K order [0..31] (lane half h gets k = 16h + s)
//     lanes 0-15 read bytes [0,32) of row (lane%16); lanes 16-31 read [32,64).
//   B stored as Bs[nt][lane][i]: lane = (n%16) + (k>=16 ? 16:0), i = k%16.
//
// Staging is two-phase (gather -> convert/store) and branch-free: B columns
// are clamped in-bounds so all global loads issue in one clause; OOB columns
// are zeroed with a VALU select after the load.
// ---------------------------------------------------------------------------
__device__ __forceinline__ float gelu_exact(float v) {
    return 0.5f * v * (1.0f + erff(v * 0.70710678118654752f));
}

// K position with 8-groups 1 and 2 swapped; not used (As rows keep natural K order)
__device__ __forceinline__ int a_kpos(int k) {
    int g  = k >> 3;                              // K group of 8
    int g2 = ((g & 1) << 1) | ((g >> 1) & 1);     // swap groups 1 and 2
    return (g2 << 3) | (k & 7);
}

__global__ __launch_bounds__(128)
void gemm_wmma_kernel(const float* __restrict__ A, const float* __restrict__ W,
                      const float* __restrict__ bias, const float* __restrict__ residual,
                      float* __restrict__ C, int M, int N, int K, int act_gelu)
{
    __shared__ __align__(64) _Float16 As[TILE_M][TILE_K];    // frag-ready rows
    __shared__ __align__(64) _Float16 Bs[4][32][16];         // [nt][lane][i]

    const int tid  = threadIdx.x;
    const int wv   = tid >> 5;
    const int lane = tid & 31;
    const int m0   = blockIdx.y * TILE_M;
    const int n0   = blockIdx.x * TILE_N;

    // this thread's B column, clamped in-bounds (branch-free OOB handling)
    const int ccol   = tid & 63;                 // col 0..63 (coalesced)
    const int cvalid = (n0 + ccol < N);
    const int ccl    = cvalid ? (n0 + ccol) : (N - 1);

    v8f acc[4] = {v8f{}, v8f{}, v8f{}, v8f{}};

    for (int k0 = 0; k0 < K; k0 += TILE_K) {
        // ---- Phase 1: gather A (8x float2) and B (2x8 floats) ----
        float2 aval[8];
        #pragma unroll
        for (int u = 0; u < 8; ++u) {
            int i = tid + u * 128;
            int r = i >> 4;                      // row 0..63
            int k = (i & 15) << 1;               // even K
            aval[u] = *(const float2*)&A[(size_t)(m0 + r) * K + (k0 + k)];
        }
        float bv0[8], bv1[8];
        #pragma unroll
        for (int u = 0; u < 8; ++u) {
            int i = tid + u * 128;
            int k = (i >> 6) << 1;               // even K (pair)
            bv0[u] = W[(size_t)(k0 + k)     * N + ccl];
            bv1[u] = W[(size_t)(k0 + k + 1) * N + ccl];
        }
        // Prefetch next K tiles (global_prefetch_b8)
        if (k0 + TILE_K < K) {
            __builtin_prefetch(&A[(size_t)(m0 + (tid >> 1)) * K + (k0 + TILE_K)], 0, 1);
            __builtin_prefetch(&W[(size_t)(k0 + TILE_K + (tid >> 2)) * N + n0], 0, 1);
        }

        // ---- Phase 2: convert to f16 and store fragment-ready ----
        #pragma unroll
        for (int u = 0; u < 8; ++u) {
            int i = tid + u * 128;
            int r = i >> 4;
            int k = (i & 15) << 1;
            v2h hv; hv[0] = (_Float16)aval[u].x; hv[1] = (_Float16)aval[u].y;
            *(v2h*)&As[r][k] = hv;       // p even -> 4B aligned
        }
        #pragma unroll
        for (int u = 0; u < 8; ++u) {
            int i = tid + u * 128;
            int k = (i >> 6) << 1;
            float w0 = cvalid ? bv0[u] : 0.f;    // VALU select, no branch
            float w1 = cvalid ? bv1[u] : 0.f;
            v2h hv; hv[0] = (_Float16)w0; hv[1] = (_Float16)w1;
            int bl = (ccol & 15) + ((k >= 16) ? 16 : 0);
            *(v2h*)&Bs[ccol >> 4][bl][k & 15] = hv;  // i even -> 4B aligned
        }
        __syncthreads();

        // ---- Fragments: contiguous 32B per lane -> 2x ds_load_b128 each ----
        const _Float16* ap = &As[wv * 16 + (lane & 15)][(lane >= 16) ? 16 : 0];
        v16h afrag = *(const v16h*)ap;

        #pragma unroll
        for (int nt = 0; nt < 4; ++nt) {
            v16h bfrag = *(const v16h*)&Bs[nt][lane][0];
            acc[nt] = __builtin_amdgcn_wmma_f32_16x16x32_f16(
                false, afrag, false, bfrag, (short)0, acc[nt], false, false);
        }
        __syncthreads();
    }

    // Epilogue: C layout -> VGPR v holds M = v (+8 for lanes 16..31), N = lane%16
    const int mrow0 = m0 + wv * 16 + ((lane >= 16) ? 8 : 0);
    #pragma unroll
    for (int nt = 0; nt < 4; ++nt) {
        int n = n0 + nt * 16 + (lane & 15);
        if (n >= N) continue;
        float bv = bias ? bias[n] : 0.0f;
        #pragma unroll
        for (int v = 0; v < 8; ++v) {
            int m = mrow0 + v;
            float val = acc[nt][v] + bv;
            if (act_gelu) val = gelu_exact(val);
            if (residual) val += residual[(size_t)m * N + n];
            C[(size_t)m * N + n] = val;
        }
    }
}

// ---------------------------------------------------------------------------
// Causal attention, online softmax. One wave32 per query row; lanes split
// the head dim (hd <= 64). grid = (B, H, T/8), block = 256 (8 waves).
// ---------------------------------------------------------------------------
__global__ __launch_bounds__(256)
void attn_kernel(const float* __restrict__ q, const float* __restrict__ k,
                 const float* __restrict__ v, float* __restrict__ y,
                 int H, int hd, int Cdim, float scale)
{
    const int wv   = threadIdx.x >> 5;
    const int lane = threadIdx.x & 31;
    const int b    = blockIdx.x;
    const int h    = blockIdx.y;
    const int t    = blockIdx.z * 8 + wv;

    const size_t rowbase = ((size_t)b * GT + t) * Cdim + (size_t)h * hd;
    const int has0 = (lane      < hd);
    const int has1 = (lane + 32 < hd);

    float q0 = has0 ? q[rowbase + lane]      * scale : 0.f;
    float q1 = has1 ? q[rowbase + lane + 32] * scale : 0.f;

    float mrun = -INFINITY, lrun = 0.f, a0 = 0.f, a1 = 0.f;

    for (int j = 0; j <= t; ++j) {
        const size_t kb = ((size_t)b * GT + j) * Cdim + (size_t)h * hd;
        float d = 0.f;
        if (has0) d += q0 * k[kb + lane];
        if (has1) d += q1 * k[kb + lane + 32];
        d = wave_sum(d);

        float mnew = fmaxf(mrun, d);
        float corr = __expf(mrun - mnew);
        float p    = __expf(d - mnew);
        lrun = lrun * corr + p;
        if (has0) a0 = a0 * corr + p * v[kb + lane];
        if (has1) a1 = a1 * corr + p * v[kb + lane + 32];
        mrun = mnew;
    }

    float inv = 1.0f / lrun;
    if (has0) y[rowbase + lane]      = a0 * inv;
    if (has1) y[rowbase + lane + 32] = a1 * inv;
}

// ---------------------------------------------------------------------------
// Host orchestration
// ---------------------------------------------------------------------------
static inline void run_gemm(const float* A, const float* W, const float* bias,
                            const float* res, float* C, int M, int N, int K,
                            int gelu, hipStream_t s)
{
    dim3 grid((N + TILE_N - 1) / TILE_N, M / TILE_M);
    gemm_wmma_kernel<<<grid, 128, 0, s>>>(A, W, bias, res, C, M, N, K, gelu);
}

extern "C" void kernel_launch(void* const* d_in, const int* in_sizes, int n_in,
                              void* d_out, int out_size, void* d_ws, size_t ws_size,
                              hipStream_t stream)
{
    (void)in_sizes; (void)n_in; (void)out_size; (void)ws_size;

    // ---- unpack inputs (setup_inputs() insertion order) ----
    const int*   idx     = (const int*)  d_in[1];
    const int*   actions = (const int*)  d_in[2];
    const float* tok_emb = (const float*)d_in[3];
    const float* pos_emb = (const float*)d_in[4];
    const float* act_emb = (const float*)d_in[5];

    // Per-block parameter tables. Block 0 (has "out") uses 18 tensors, rest 16.
    struct BP { const float *ln1g,*ln1b,*ln2g,*ln2b,*qW,*qb,*kW,*kb,*vW,*vb,
                            *pW,*pb,*m1W,*m1b,*m2W,*m2b,*oW,*ob; };
    BP bp[4];
    {
        int base = 6;
        for (int blk = 0; blk < 4; ++blk) {
            BP& p = bp[blk];
            p.ln1g = (const float*)d_in[base + 0];
            p.ln1b = (const float*)d_in[base + 1];
            p.ln2g = (const float*)d_in[base + 2];
            p.ln2b = (const float*)d_in[base + 3];
            p.qW   = (const float*)d_in[base + 4];
            p.qb   = (const float*)d_in[base + 5];
            p.kW   = (const float*)d_in[base + 6];
            p.kb   = (const float*)d_in[base + 7];
            p.vW   = (const float*)d_in[base + 8];
            p.vb   = (const float*)d_in[base + 9];
            p.pW   = (const float*)d_in[base + 10];
            p.pb   = (const float*)d_in[base + 11];
            p.m1W  = (const float*)d_in[base + 12];
            p.m1b  = (const float*)d_in[base + 13];
            p.m2W  = (const float*)d_in[base + 14];
            p.m2b  = (const float*)d_in[base + 15];
            if (blk == 0) {             // only block 0 has the "out" projection
                p.oW = (const float*)d_in[base + 16];
                p.ob = (const float*)d_in[base + 17];
                base += 18;
            } else {
                p.oW = nullptr; p.ob = nullptr;
                base += 16;
            }
        }
        // base == 72 here: lnf_g, lnf_b, W_logit, W_dev follow.
    }
    const float* lnf_g   = (const float*)d_in[72];
    const float* lnf_b   = (const float*)d_in[73];
    const float* W_logit = (const float*)d_in[74];
    const float* W_dev   = (const float*)d_in[75];

    // ---- workspace carve (floats) ----
    const size_t MD  = (size_t)GM * GD;       // 4096*1024
    const size_t MH  = (size_t)GM * 4 * GD;   // mlp hidden (block 0 worst case)
    float* ws   = (float*)d_ws;
    float* bufX  = ws;            ws += MD;   // current activations
    float* bufY  = ws;            ws += MD;   // alt activations (after "out")
    float* bufLN = ws;            ws += MD;   // LN output / attention output
    float* bufQ  = ws;            ws += MD;
    float* bufK  = ws;            ws += MD;
    float* bufV  = ws;            ws += MD;
    float* bufH  = ws;            ws += MH;   // MLP hidden

    // ---- embedding ----
    embed_kernel<<<GM, 256, 0, stream>>>(idx, actions, tok_emb, pos_emb,
                                         act_emb, bufX);

    // ---- transformer blocks ----
    // layer dims per reference: block0 (1024 -> 96), blocks 1-3 (96 -> 96)
    int din = GD;
    float* xcur = bufX;
    float* xalt = bufY;
    for (int blk = 0; blk < 4; ++blk) {
        const BP& p = bp[blk];
        const int hd = din / GH;
        const float scale = 1.0f / sqrtf((float)hd);

        // ln1
        ln_kernel<<<GM, 256, 0, stream>>>(xcur, p.ln1g, p.ln1b, bufLN, din);
        // q, k, v
        run_gemm(bufLN, p.qW, p.qb, nullptr, bufQ, GM, din, din, 0, stream);
        run_gemm(bufLN, p.kW, p.kb, nullptr, bufK, GM, din, din, 0, stream);
        run_gemm(bufLN, p.vW, p.vb, nullptr, bufV, GM, din, din, 0, stream);
        // attention -> bufLN (ln output no longer needed)
        {
            dim3 agrid(GB, GH, GT / 8);
            attn_kernel<<<agrid, 256, 0, stream>>>(bufQ, bufK, bufV, bufLN,
                                                   GH, hd, din, scale);
        }
        // x = attn @ proj + b + x   (in-place residual is per-element safe)
        run_gemm(bufLN, p.pW, p.pb, xcur, xcur, GM, din, din, 0, stream);
        // ln2
        ln_kernel<<<GM, 256, 0, stream>>>(xcur, p.ln2g, p.ln2b, bufLN, din);
        // mlp: h = gelu(ln @ W1 + b1); x = h @ W2 + b2 + x
        run_gemm(bufLN, p.m1W, p.m1b, nullptr, bufH, GM, 4 * din, din, 1, stream);
        run_gemm(bufH,  p.m2W, p.m2b, xcur, xcur, GM, din, 4 * din, 0, stream);
        // optional down-projection (block 0: 1024 -> 96)
        if (p.oW) {
            const int dout = 96;
            run_gemm(xcur, p.oW, p.ob, nullptr, xalt, GM, dout, din, 0, stream);
            float* t = xcur; xcur = xalt; xalt = t;
            din = dout;
        }
    }

    // ---- final LN + heads ----
    ln_kernel<<<GM, 256, 0, stream>>>(xcur, lnf_g, lnf_b, bufLN, din);
    float* logits = (float*)d_out;
    float* dev    = logits + (size_t)GM * GVOCAB;
    run_gemm(bufLN, W_logit, nullptr, nullptr, logits, GM, GVOCAB, din, 0, stream);
    run_gemm(bufLN, W_dev,   nullptr, nullptr, dev,    GM, GVOCAB, din, 0, stream);
}
